// DeepLSTM_69337952027272
// MI455X (gfx1250) — compile-verified, matrix operand order checked by reading, not run on hardware
//
#include <hip/hip_runtime.h>

// Problem dims
#define B_   32
#define S_   512
#define IN_  512
#define H_   1024
#define L_   4
#define G3H  (3 * H_)

// Persistent recurrent kernel config: 64 blocks x 64 threads (2 waves).
// Each block owns a 16-wide slice of H (3 gates -> 48 gate rows in LDS).
#define NBLK 64
#define NTHR 64

typedef __bf16 bf16_t;
typedef bf16_t v16bf __attribute__((ext_vector_type(16)));
typedef float  v8f   __attribute__((ext_vector_type(8)));
typedef unsigned int u32x4 __attribute__((ext_vector_type(4)));

union Frag { v16bf v; u32x4 q[2]; };

// LDS weight tile layout: [kt][gate(3)][row(16)][k(32)] bf16.
// Per k-step all 6 B-fragment ds_loads share one base address with
// immediate offsets {0,32,1024,1056,2048,2080} bytes.
#define TILE_US   (3 * 16 * 32)   // ushorts per k-tile block (3072 B)
#define GATE_US   (16 * 32)       // 512 ushorts = 1024 B

__device__ __forceinline__ unsigned short f32_to_bf16(float f) {
  union { float f; unsigned u; } c; c.f = f;
  unsigned u = c.u;
  unsigned r = (u + 0x7FFFu + ((u >> 16) & 1u)) >> 16; // RNE
  return (unsigned short)r;
}

__device__ __forceinline__ float fsig(float x) {
  return 1.0f / (1.0f + __expf(-x));
}
__device__ __forceinline__ float ftanh(float x) {
  float ax = fabsf(x);
  float e = __expf(-2.0f * ax);
  float t = (1.0f - e) / (1.0f + e);
  return x < 0.0f ? -t : t;
}

// ---- fp32 -> bf16 bulk convert ----
__global__ void cvt_f32_bf16(const float* __restrict__ src,
                             unsigned short* __restrict__ dst, int n) {
  int i = blockIdx.x * blockDim.x + threadIdx.x;
  int stride = gridDim.x * blockDim.x;
  for (; i < n; i += stride) dst[i] = f32_to_bf16(src[i]);
}

__global__ void init_barrier(unsigned* bar) { bar[0] = 0u; bar[1] = 0u; }

// Device-wide sense-reversing barrier (all NBLK blocks co-resident:
// persistent kernel, 64 blocks of 64 threads, <=196KB LDS each).
__device__ __forceinline__ void device_barrier(unsigned* bar) {
  __threadfence();
  __syncthreads();
  if (threadIdx.x == 0) {
    volatile unsigned* vb = (volatile unsigned*)bar;
    unsigned gen = vb[1];
    unsigned arr = atomicAdd(&bar[0], 1u);
    if (arr == NBLK - 1) {
      vb[0] = 0u;
      __threadfence();
      atomicAdd(&bar[1], 1u);
    } else {
      while (vb[1] == gen) __builtin_amdgcn_s_sleep(2);
    }
  }
  __syncthreads();
  __threadfence(); // acquire: don't read stale near-cache lines
}

// ---- persistent per-layer LSTM kernel ----
// gates = [x_t | h] @ [W_ih | W_hh]^T + b ; c += sig(i)*tanh(g); h = sig(o)*tanh(c)
__global__ void lstm_layer_kernel(
    const unsigned short* __restrict__ xin,  // bf16 [B, S, Din]
    int Din,
    const unsigned short* __restrict__ wih,  // bf16 [3H, Din] row-major
    const unsigned short* __restrict__ whh,  // bf16 [3H, H]   row-major
    const float* __restrict__ bias,          // f32  [3H]
    unsigned short* __restrict__ hbuf,       // bf16 [2, B, H] double-buffered state
    unsigned* __restrict__ bar,
    unsigned short* __restrict__ yout_bf,    // bf16 [B, S, H] (next layer input) or null
    float* __restrict__ yout_f32,            // f32  [B, S, H] (final layer) or null
    float* __restrict__ hfin,                // f32 [B, H]
    float* __restrict__ cfin)                // f32 [B, H]
{
  extern __shared__ unsigned short wlds[];   // [kTiles][3][16][32]
  const int Ktot  = Din + H_;
  const int chunk = blockIdx.x * 16;         // base h-column of this block
  const int wave  = threadIdx.x >> 5;        // 0..1 -> 16-row M tile
  const int lane  = threadIdx.x & 31;
  const int nl    = lane & 15;
  const int hi    = lane >> 4;

  // Stage this block's 48 gate rows into the k-tile-interleaved LDS layout.
  for (int r = 0; r < 48; ++r) {
    const int g  = r >> 4;
    const int n  = r & 15;
    const int gr = g * H_ + chunk + n;                 // global gate row
    const unsigned short* s0 = wih + (size_t)gr * Din;
    for (int k = threadIdx.x * 8; k < Din; k += NTHR * 8) {
      const int kt = k >> 5;
      *(u32x4*)(wlds + (size_t)kt * TILE_US + g * GATE_US + n * 32 + (k & 31)) =
          *(const u32x4*)(s0 + k);
    }
    const unsigned short* s1 = whh + (size_t)gr * H_;
    for (int k2 = threadIdx.x * 8; k2 < H_; k2 += NTHR * 8) {
      const int k  = Din + k2;
      const int kt = k >> 5;
      *(u32x4*)(wlds + (size_t)kt * TILE_US + g * GATE_US + n * 32 + (k & 31)) =
          *(const u32x4*)(s1 + k2);
    }
  }

  // Zero parity-0 h state for this block's columns.
  for (int idx = threadIdx.x; idx < B_ * 16; idx += NTHR) {
    int b = idx >> 4, cc = idx & 15;
    hbuf[(size_t)b * H_ + chunk + cc] = 0;
  }

  // Per-lane gate biases (same for all 8 rows this lane owns).
  const float bias_i = bias[0 * H_ + chunk + nl];
  const float bias_g = bias[1 * H_ + chunk + nl];
  const float bias_o = bias[2 * H_ + chunk + nl];

  // CEC cell state in registers (C layout: rows wave*16+hi*8+j, col chunk+nl).
  float creg[8];
#pragma unroll
  for (int j = 0; j < 8; ++j) creg[j] = 0.0f;

  device_barrier(bar); // LDS staged + h zeroed everywhere before first step

  // A-fragment: lane m+16*hi holds K[hi*8..+8] and K[16+hi*8..+8] of row m.
  const int mbase   = wave * 16 + nl;
  const int rowc    = wave * 16 + hi * 8;
  const int col     = chunk + nl;
  const int kSplit  = Din >> 5;          // k-tiles in x part
  const int kTilesH = H_ >> 5;           // k-tiles in h part
  const unsigned short* bstart = wlds + nl * 32 + hi * 8;

  for (int t = 0; t < S_; ++t) {
    v8f acc0, acc1, acc2;
#pragma unroll
    for (int j = 0; j < 8; ++j) { acc0[j] = bias_i; acc1[j] = bias_g; acc2[j] = bias_o; }

    const unsigned short* ax = xin + ((size_t)mbase * S_ + t) * Din + hi * 8;
    const unsigned short* ah = hbuf + (size_t)(t & 1) * (B_ * H_) + (size_t)mbase * H_ + hi * 8;
    const unsigned short* bptr = bstart;

    // Pull next timestep's x row toward the WGP while we compute.
    if (t + 1 < S_)
      __builtin_prefetch(xin + ((size_t)mbase * S_ + (t + 1)) * Din, 0, 1);

    // ---- x part: gates += x_t @ W_ih^T ----
#pragma unroll 4
    for (int kt = 0; kt < kSplit; ++kt) {
      Frag a, b0, b1, b2;
      a.q[0]  = *(const u32x4*)(ax);
      a.q[1]  = *(const u32x4*)(ax + 16);
      b0.q[0] = *(const u32x4*)(bptr);
      b0.q[1] = *(const u32x4*)(bptr + 16);
      b1.q[0] = *(const u32x4*)(bptr + GATE_US);
      b1.q[1] = *(const u32x4*)(bptr + GATE_US + 16);
      b2.q[0] = *(const u32x4*)(bptr + 2 * GATE_US);
      b2.q[1] = *(const u32x4*)(bptr + 2 * GATE_US + 16);
      acc0 = __builtin_amdgcn_wmma_f32_16x16x32_bf16(false, a.v, false, b0.v, (short)0, acc0, false, false);
      acc1 = __builtin_amdgcn_wmma_f32_16x16x32_bf16(false, a.v, false, b1.v, (short)0, acc1, false, false);
      acc2 = __builtin_amdgcn_wmma_f32_16x16x32_bf16(false, a.v, false, b2.v, (short)0, acc2, false, false);
      ax   += 32;
      bptr += TILE_US;
    }

    // ---- h part: gates += h_t @ W_hh^T (critical path, all operands on-chip) ----
#pragma unroll 4
    for (int kt = 0; kt < kTilesH; ++kt) {
      Frag a, b0, b1, b2;
      a.q[0]  = *(const u32x4*)(ah);
      a.q[1]  = *(const u32x4*)(ah + 16);
      b0.q[0] = *(const u32x4*)(bptr);
      b0.q[1] = *(const u32x4*)(bptr + 16);
      b1.q[0] = *(const u32x4*)(bptr + GATE_US);
      b1.q[1] = *(const u32x4*)(bptr + GATE_US + 16);
      b2.q[0] = *(const u32x4*)(bptr + 2 * GATE_US);
      b2.q[1] = *(const u32x4*)(bptr + 2 * GATE_US + 16);
      acc0 = __builtin_amdgcn_wmma_f32_16x16x32_bf16(false, a.v, false, b0.v, (short)0, acc0, false, false);
      acc1 = __builtin_amdgcn_wmma_f32_16x16x32_bf16(false, a.v, false, b1.v, (short)0, acc1, false, false);
      acc2 = __builtin_amdgcn_wmma_f32_16x16x32_bf16(false, a.v, false, b2.v, (short)0, acc2, false, false);
      ah   += 32;
      bptr += TILE_US;
    }

    // ---- elementwise CEC update; write h_{t+1} into the other parity ----
    unsigned short* hw = hbuf + (size_t)((t + 1) & 1) * (B_ * H_);
#pragma unroll
    for (int j = 0; j < 8; ++j) {
      const int row = rowc + j;
      const float ig = fsig(acc0[j]);
      const float gg = ftanh(acc1[j]);
      const float c  = creg[j] + ig * gg;   // no forget gate (CEC)
      creg[j] = c;
      const float hv = fsig(acc2[j]) * ftanh(c);

      hw[(size_t)row * H_ + col] = f32_to_bf16(hv);
      const size_t oidx = ((size_t)row * S_ + t) * H_ + col;
      if (yout_f32) yout_f32[oidx] = hv;
      if (yout_bf)  yout_bf[oidx]  = f32_to_bf16(hv);
      if (t == S_ - 1) {
        hfin[(size_t)row * H_ + col] = hv;
        cfin[(size_t)row * H_ + col] = c;
      }
    }
    device_barrier(bar); // h_{t+1} visible everywhere before step t+1
  }
}

extern "C" void kernel_launch(void* const* d_in, const int* in_sizes, int n_in,
                              void* d_out, int out_size, void* d_ws, size_t ws_size,
                              hipStream_t stream) {
  const float* x    = (const float*)d_in[0];
  const float* wih0 = (const float*)d_in[1];
  const float* whh0 = (const float*)d_in[2];
  const float* b0   = (const float*)d_in[3];
  const float* wihR = (const float*)d_in[4];
  const float* whhR = (const float*)d_in[5];
  const float* bR   = (const float*)d_in[6];
  float* out = (float*)d_out;

  // Workspace layout (bytes): [barrier 64B][bf16 arrays ...] (~131 MB total)
  char* ws = (char*)d_ws;
  unsigned* bar = (unsigned*)ws;
  size_t off = 64;
  auto alloc_us = [&](size_t n) {
    unsigned short* p = (unsigned short*)(ws + off); off += n * 2; return p;
  };
  unsigned short* xbf   = alloc_us((size_t)B_ * S_ * IN_);
  unsigned short* wih0b = alloc_us((size_t)G3H * IN_);
  unsigned short* whh0b = alloc_us((size_t)G3H * H_);
  unsigned short* wihRb = alloc_us((size_t)(L_ - 1) * G3H * H_);
  unsigned short* whhRb = alloc_us((size_t)(L_ - 1) * G3H * H_);
  unsigned short* seqA  = alloc_us((size_t)B_ * S_ * H_);
  unsigned short* seqB  = alloc_us((size_t)B_ * S_ * H_);
  unsigned short* hbuf  = alloc_us((size_t)2 * B_ * H_);
  (void)ws_size; (void)in_sizes; (void)n_in; (void)out_size;

  auto cvt = [&](const float* s, unsigned short* d, size_t n) {
    int blocks = (int)((n + 255) / 256);
    hipLaunchKernelGGL(cvt_f32_bf16, dim3(blocks), dim3(256), 0, stream, s, d, (int)n);
  };
  cvt(x,    xbf,   (size_t)B_ * S_ * IN_);
  cvt(wih0, wih0b, (size_t)G3H * IN_);
  cvt(whh0, whh0b, (size_t)G3H * H_);
  cvt(wihR, wihRb, (size_t)(L_ - 1) * G3H * H_);
  cvt(whhR, whhRb, (size_t)(L_ - 1) * G3H * H_);
  hipLaunchKernelGGL(init_barrier, dim3(1), dim3(1), 0, stream, bar);

  float* hfin = out + (size_t)B_ * S_ * H_;
  float* cfin = hfin + (size_t)L_ * B_ * H_;
  const size_t wstride = (size_t)G3H * H_;
  const unsigned lds0 = 48 * (IN_ + H_) * 2;  // 147456 B
  const unsigned lds1 = 48 * (2 * H_) * 2;    // 196608 B (<= 320KB WGP LDS)

  lstm_layer_kernel<<<NBLK, NTHR, lds0, stream>>>(
      xbf, IN_, wih0b, whh0b, b0, hbuf, bar, seqA, nullptr,
      hfin + 0 * (size_t)B_ * H_, cfin + 0 * (size_t)B_ * H_);
  lstm_layer_kernel<<<NBLK, NTHR, lds1, stream>>>(
      seqA, H_, wihRb + 0 * wstride, whhRb + 0 * wstride, bR + 0 * G3H,
      hbuf, bar, seqB, nullptr,
      hfin + 1 * (size_t)B_ * H_, cfin + 1 * (size_t)B_ * H_);
  lstm_layer_kernel<<<NBLK, NTHR, lds1, stream>>>(
      seqB, H_, wihRb + 1 * wstride, whhRb + 1 * wstride, bR + 1 * G3H,
      hbuf, bar, seqA, nullptr,
      hfin + 2 * (size_t)B_ * H_, cfin + 2 * (size_t)B_ * H_);
  lstm_layer_kernel<<<NBLK, NTHR, lds1, stream>>>(
      seqA, H_, wihRb + 2 * wstride, whhRb + 2 * wstride, bR + 2 * G3H,
      hbuf, bar, nullptr, out,
      hfin + 3 * (size_t)B_ * H_, cfin + 3 * (size_t)B_ * H_);
}
